// MassiveBinaryLayer_37245956391175
// MI455X (gfx1250) — 8-bit matrix operand order repaired and checked by reading, compiled, not run on hardware
//
#include <hip/hip_runtime.h>
#include <hip/hip_bf16.h>
#include <stdint.h>

typedef __attribute__((ext_vector_type(16))) int          v16i;
typedef __attribute__((ext_vector_type(8)))  float        v8f;
typedef __attribute__((ext_vector_type(4)))  unsigned int v4u;
typedef __attribute__((ext_vector_type(8)))  int          v8i_t;
typedef __attribute__((ext_vector_type(4)))  int          v4i_t;

#define BATCH 4096
#define INSZ  4096
#define OUTSZ 8192
#define THRESH 0.5f

#define BM 128
#define BN 64
#define BK 128
#define LDT 144
#define NK  (INSZ / BK)

__device__ __forceinline__ unsigned int f32_to_e4m3(float f) {
    union { float f; unsigned int u; } v; v.f = f;
    const unsigned int u    = v.u;
    const unsigned int sign = (u >> 24) & 0x80u;
    const int          e    = (int)((u >> 23) & 0xFF) - 127;
    const unsigned int m    = u & 0x7FFFFFu;
    if (e < -9) return sign;
    if (e > 8)  return sign | 0x7Eu;
    int e8 = e + 7;
    if (e8 >= 1) {
        unsigned int keep = m >> 20;
        unsigned int rest = m & 0xFFFFFu;
        keep += (rest > 0x80000u) || (rest == 0x80000u && (keep & 1u));
        if (keep == 8u) { keep = 0u; ++e8; if (e8 > 15) return sign | 0x7Eu; }
        return sign | ((unsigned int)e8 << 3) | keep;
    } else {
        const int shift = 21 - e8;
        const unsigned int full = 0x800000u | m;
        unsigned int keep = full >> shift;
        const unsigned int rem  = full & ((1u << shift) - 1u);
        const unsigned int half = 1u << (shift - 1);
        keep += (rem > half) || (rem == half && (keep & 1u));
        return sign | keep;
    }
}

__global__ void convert_x_fp8(const float* __restrict__ x,
                              unsigned char* __restrict__ xq, int n) {
    const int i = (blockIdx.x * blockDim.x + threadIdx.x) * 4;
    if (i >= n) return;
    const float4 f = *(const float4*)(x + i);
    unsigned int p =  f32_to_e4m3(f.x)
                   | (f32_to_e4m3(f.y) << 8)
                   | (f32_to_e4m3(f.z) << 16)
                   | (f32_to_e4m3(f.w) << 24);
    *(unsigned int*)(xq + i) = p;
}

__global__ void convert_w_fp8(const float* __restrict__ w,
                              unsigned char* __restrict__ wq, int n) {
    const int i = (blockIdx.x * blockDim.x + threadIdx.x) * 4;
    if (i >= n) return;
    const float4 f = *(const float4*)(w + i);
    unsigned int p =  (f.x > 0.0f ? 0x38u : 0u)
                   | ((f.y > 0.0f ? 0x38u : 0u) << 8)
                   | ((f.z > 0.0f ? 0x38u : 0u) << 16)
                   | ((f.w > 0.0f ? 0x38u : 0u) << 24);
    *(unsigned int*)(wq + i) = p;
}

__device__ __forceinline__ void tdm_load_tile(
    const unsigned char* gptr, unsigned lds_addr,
    unsigned tensor_d0, unsigned tensor_d1,
    unsigned tile_d0, unsigned tile_d1, unsigned stride0)
{
    const unsigned long long ga = (unsigned long long)(uintptr_t)gptr;
    v4u g0;
    g0.x = 1u;
    g0.y = lds_addr;
    g0.z = (unsigned)(ga & 0xFFFFFFFFull);
    g0.w = (unsigned)((ga >> 32) & 0x01FFFFFFull)
         | (2u << 30);
    v8i_t g1;
    g1[0] = (int)((1u << 20) | (4u << 22) | (3u << 25));
    g1[1] = (int)(tensor_d0 << 16);
    g1[2] = (int)((tensor_d0 >> 16) | (tensor_d1 << 16));
    g1[3] = (int)((tensor_d1 >> 16) | (tile_d0 << 16));
    g1[4] = (int)tile_d1;
    g1[5] = (int)stride0;
    g1[6] = 0;
    g1[7] = 0;
    const v4i_t gz4 = {0, 0, 0, 0};
    const v8i_t gz8 = {0, 0, 0, 0, 0, 0, 0, 0};
    __builtin_amdgcn_tensor_load_to_lds(g0, g1, gz4, gz4, gz8, 0);
}

__global__ void __launch_bounds__(256)
binary_gemm_fp8(const unsigned char* __restrict__ Aq,
                const unsigned char* __restrict__ Bq,
                const float* __restrict__ bias,
                float* __restrict__ out)
{
    __shared__ __align__(16) unsigned char lA[2][BM * LDT];
    __shared__ __align__(16) unsigned char lB[2][BN * LDT];

    const int tid  = threadIdx.x;
    const int lane = tid & 31;
    const int wave = tid >> 5;
    const int mw   = wave >> 1;
    const int nw   = wave & 1;
    const int lrow = lane & 15;
    const int lhi  = lane >> 4;

    const int m0 = blockIdx.y * BM;
    const int n0 = blockIdx.x * BN;

    const unsigned baseA = (unsigned)(uintptr_t)&lA[0][0];
    const unsigned baseB = (unsigned)(uintptr_t)&lB[0][0];

    v8f acc[2][2];
    const v8f vzero = {0.f, 0.f, 0.f, 0.f, 0.f, 0.f, 0.f, 0.f};
#pragma unroll
    for (int i = 0; i < 2; ++i)
#pragma unroll
        for (int j = 0; j < 2; ++j) acc[i][j] = vzero;

    if (wave == 0) {
#pragma unroll
        for (int p = 0; p < 2; ++p) {
            tdm_load_tile(Aq + (size_t)m0 * INSZ + p * BK,
                          baseA + (unsigned)p * (BM * LDT),
                          INSZ, BATCH, BK, BM, INSZ);
            tdm_load_tile(Bq + (size_t)n0 * INSZ + p * BK,
                          baseB + (unsigned)p * (BN * LDT),
                          INSZ, OUTSZ, BK, BN, INSZ);
        }
    }

    for (int kt = 0; kt < NK; ++kt) {
        const int cur = kt & 1;

        if (wave == 0) {
            if (kt + 1 < NK) __builtin_amdgcn_s_wait_tensorcnt(2);
            else             __builtin_amdgcn_s_wait_tensorcnt(0);
        }
        __syncthreads();

        union { v16i v; unsigned long long q[8]; } afrag[2];
        union { v16i v; uint4 x[4]; }              bfrag[2];
#pragma unroll
        for (int mt = 0; mt < 2; ++mt) {
            const int row = mw * 32 + mt * 16 + lrow;
#pragma unroll
            for (int j = 0; j < 8; ++j)
                afrag[mt].q[j] = *(const unsigned long long*)
                    &lA[cur][row * LDT + (lhi << 4) + ((j >> 1) << 5) + ((j & 1) << 3)];
        }
#pragma unroll
        for (int nt = 0; nt < 2; ++nt) {
            const int col = nw * 32 + nt * 16 + lrow;
#pragma unroll
            for (int j = 0; j < 4; ++j)
                bfrag[nt].x[j] = *(const uint4*)
                    &lB[cur][col * LDT + (lhi << 4) + (j << 5)];
        }

#pragma unroll
        for (int mt = 0; mt < 2; ++mt)
#pragma unroll
            for (int nt = 0; nt < 2; ++nt)
                acc[mt][nt] = __builtin_amdgcn_wmma_f32_16x16x128_fp8_fp8(
                    afrag[mt].v, bfrag[nt].v, (short)0, acc[mt][nt], false, false);

        __syncthreads();

        if (wave == 0 && kt + 2 < NK) {
            const int k2 = (kt + 2) * BK;
            tdm_load_tile(Aq + (size_t)m0 * INSZ + k2,
                          baseA + (unsigned)cur * (BM * LDT),
                          INSZ, BATCH, BK, BM, INSZ);
            tdm_load_tile(Bq + (size_t)n0 * INSZ + k2,
                          baseB + (unsigned)cur * (BN * LDT),
                          INSZ, OUTSZ, BK, BN, INSZ);
        }
    }

#pragma unroll
    for (int nt = 0; nt < 2; ++nt) {
        const int col = n0 + nw * 32 + nt * 16 + lrow;
        const float bv = bias[col];
#pragma unroll
        for (int mt = 0; mt < 2; ++mt) {
            const int rbase = m0 + mw * 32 + mt * 16 + lhi * 8;
#pragma unroll
            for (int e = 0; e < 8; ++e) {
                const float s = acc[mt][nt][e] + bv;
                out[(size_t)(rbase + e) * OUTSZ + col] = (s > THRESH) ? 1.0f : 0.0f;
            }
        }
    }
}

extern "C" void kernel_launch(void* const* d_in, const int* in_sizes, int n_in,
                              void* d_out, int out_size, void* d_ws, size_t ws_size,
                              hipStream_t stream) {
    const float* x    = (const float*)d_in[0];
    const float* w    = (const float*)d_in[1];
    const float* bias = (const float*)d_in[2];
    float*       out  = (float*)d_out;

    unsigned char* xq = (unsigned char*)d_ws;
    unsigned char* wq = xq + (size_t)BATCH * INSZ;

    const int nx = BATCH * INSZ;
    const int nw = OUTSZ * INSZ;
    convert_x_fp8<<<nx / (256 * 4), 256, 0, stream>>>(x, xq, nx);
    convert_w_fp8<<<nw / (256 * 4), 256, 0, stream>>>(w, wq, nw);

    dim3 grid(OUTSZ / BN, BATCH / BM);
    binary_gemm_fp8<<<grid, 256, 0, stream>>>(xq, wq, bias, out);
}
